// SpatialLiDAREncoder_33354716020752
// MI455X (gfx1250) — compile-verified, matrix operand order checked by reading, not run on hardware
//
#include <hip/hip_runtime.h>
#include <hip/hip_bf16.h>

typedef __attribute__((ext_vector_type(16))) _Float16 v16h;
typedef __attribute__((ext_vector_type(8)))  _Float16 v8h;
typedef __attribute__((ext_vector_type(8)))  float    v8f;

#define EPSF 1e-5f

// Fragment region sizes in halfs (each fragment = 32 lanes * 16 halfs = 512)
#define F0_HALFS (4  * 512)   // layer0: 4 n-tiles x 1 k-tile
#define F1_HALFS (16 * 512)   // layer1: 8 n-tiles x 2 k-tiles
#define F2_HALFS (32 * 512)   // layer2: 8 n-tiles x 4 k-tiles
#define FT_HALFS (F0_HALFS + F1_HALFS + F2_HALFS)  // 26624 halfs = 53248 B

#define HW   (128 * 128)          // 16384
#define BVOL (128 * HW)           // 2097152 (FEAT*H*W per batch)
#define MT   2                    // M-tiles (16 points each) per wave

// K position inside a 32-wide chunk for (lane, half-slot j) per the CDNA5
// 16-bit A/B fragment layout: VGPR v=j>>1, half h=j&1,
// k = (v>=4 ? 16:0) + (lane>=16 ? 8:0) + 2*(v&3) + h
__device__ __forceinline__ int kmap(int lane, int j) {
  int v = j >> 1, h = j & 1;
  return ((v & 4) ? 16 : 0) + ((lane & 16) ? 8 : 0) + ((v & 3) << 1) + h;
}

// ---------------------------------------------------------------------------
// Prep: fold BatchNorm into weights/biases, emit f16 weights pre-swizzled into
// per-lane WMMA B-fragment layout, plus f32 effective biases.
// ---------------------------------------------------------------------------
__global__ void prep_kernel(
    const float* __restrict__ w0, const float* __restrict__ b0,
    const float* __restrict__ g0, const float* __restrict__ be0,
    const float* __restrict__ m0, const float* __restrict__ vv0,
    const float* __restrict__ w1, const float* __restrict__ b1,
    const float* __restrict__ g1, const float* __restrict__ be1,
    const float* __restrict__ m1, const float* __restrict__ vv1,
    const float* __restrict__ w2, const float* __restrict__ b2,
    const float* __restrict__ g2, const float* __restrict__ be2,
    const float* __restrict__ m2, const float* __restrict__ vv2,
    _Float16* __restrict__ frags, float* __restrict__ biases) {
  int idx = blockIdx.x * blockDim.x + threadIdx.x;
  if (idx < FT_HALFS) {
    int rel, ntile, ktile, cin;
    const float *w, *g, *v;
    if (idx < F0_HALFS) {
      rel = idx; int f = rel >> 9; ntile = f; ktile = 0;
      cin = 4;   w = w0; g = g0; v = vv0;
    } else if (idx < F0_HALFS + F1_HALFS) {
      rel = idx - F0_HALFS; int f = rel >> 9; ntile = f >> 1; ktile = f & 1;
      cin = 64;  w = w1; g = g1; v = vv1;
    } else {
      rel = idx - F0_HALFS - F1_HALFS; int f = rel >> 9; ntile = f >> 2; ktile = f & 3;
      cin = 128; w = w2; g = g2; v = vv2;
    }
    int within = rel & 511;
    int lane = within >> 4;
    int j    = within & 15;
    int k = ktile * 32 + kmap(lane, j);
    int n = ntile * 16 + (lane & 15);
    float val = 0.f;
    if (k < cin) {
      float scale = g[n] * rsqrtf(v[n] + EPSF);
      val = w[n * cin + k] * scale;            // B[k][n] = W_eff[n][k]
    }
    frags[idx] = (_Float16)val;
  } else if (idx < FT_HALFS + 320) {
    int bi = idx - FT_HALFS;
    const float *b, *g, *be, *m, *v; int n;
    if (bi < 64)       { n = bi;       b = b0; g = g0; be = be0; m = m0; v = vv0; }
    else if (bi < 192) { n = bi - 64;  b = b1; g = g1; be = be1; m = m1; v = vv1; }
    else               { n = bi - 192; b = b2; g = g2; be = be2; m = m2; v = vv2; }
    float scale = g[n] * rsqrtf(v[n] + EPSF);
    biases[bi] = (b[n] - m[n]) * scale + be[n];
  }
}

// ---------------------------------------------------------------------------
// Zero-init the output BEV grid (harness poisons d_out with 0xAA).
// ---------------------------------------------------------------------------
__global__ void zero_kernel(float4* __restrict__ out) {
  size_t i = (size_t)blockIdx.x * blockDim.x + threadIdx.x;
  out[i] = float4{0.f, 0.f, 0.f, 0.f};
}

// ---------------------------------------------------------------------------
// Main: per-wave 32-point macro-tile (2 M-tiles) through 3 WMMA layers
// + scatter-max. B fragments are loaded once per (nt,kt) and reused across
// both M-tiles to halve weight-cache traffic per point.
// ---------------------------------------------------------------------------
__global__ __launch_bounds__(256) void pillar_kernel(
    const float* __restrict__ points,
    const _Float16* __restrict__ frags,
    const float* __restrict__ biases,
    float* __restrict__ out) {
  __shared__ __align__(16) _Float16 stage[8][16][128];  // 4 KB per wave, reused per M-tile
  __shared__ int sidx[8][32];

  const int tid  = threadIdx.x;
  const int wv   = tid >> 5;
  const int lane = tid & 31;
  const int tp   = blockIdx.x * 8 + wv;     // 0..14999 (120000/32 = 3750 pairs/batch)
  const int bb   = tp / 3750;
  const int n0   = (tp % 3750) * 32;

  const int nl    = lane & 15;
  const int mbase = (lane & 16) ? 8 : 0;
  const int koff  = (lane & 16) ? 8 : 0;

  // ---- build A0 frags (16x32 f16, K=0..3 live) + per-point cell indices
  v16h a0[MT];
  #pragma unroll
  for (int m = 0; m < MT; ++m) a0[m] = (v16h){};
  if (lane < 16) {
    #pragma unroll
    for (int m = 0; m < MT; ++m) {
      const float4 p =
          *(const float4*)(points + ((size_t)(bb * 120000 + n0 + m * 16 + lane)) * 4);
      a0[m][0] = (_Float16)p.x; a0[m][1] = (_Float16)p.y;
      a0[m][2] = (_Float16)p.z; a0[m][3] = (_Float16)p.w;
      float xn = (p.x + 50.f) * 0.01f;
      float yn = (p.y + 50.f) * 0.01f;
      bool valid = (xn >= 0.f) && (xn <= 1.f) && (yn >= 0.f) && (yn <= 1.f);
      int gx = (int)(xn * 127.f); gx = gx < 0 ? 0 : (gx > 127 ? 127 : gx);
      int gy = (int)(yn * 127.f); gy = gy < 0 ? 0 : (gy > 127 ? 127 : gy);
      sidx[wv][m * 16 + lane] = valid ? (bb * BVOL + gy * 128 + gx) : -1;
    }
  }

  const v8f z = {};
  v8f acc[MT][8];

  // ---- layer 0: 16x32 @ 32x64 (4 nt x MT WMMAs); B loaded once per nt
  #pragma unroll
  for (int nt = 0; nt < 4; ++nt) {
    v16h bf = *(const v16h*)(frags + nt * 512 + lane * 16);
    #pragma unroll
    for (int m = 0; m < MT; ++m)
      acc[m][nt] = __builtin_amdgcn_wmma_f32_16x16x32_f16(false, a0[m], false, bf,
                                                          (short)0, z, false, false);
  }

  // stage h0 per M-tile, immediately reload next-layer A frags (stage reused)
  v16h a1[MT][2];
  #pragma unroll
  for (int m = 0; m < MT; ++m) {
    #pragma unroll
    for (int nt = 0; nt < 4; ++nt) {
      float bias = biases[nt * 16 + nl];
      int nn = nt * 16 + nl;
      #pragma unroll
      for (int r = 0; r < 8; ++r) {
        float val = acc[m][nt][r] + bias;
        val = val > 0.f ? val : 0.f;
        stage[wv][mbase + r][nn] = (_Float16)val;
      }
    }
    asm volatile("s_wait_dscnt 0" ::: "memory");
    #pragma unroll
    for (int kt = 0; kt < 2; ++kt) {
      v8h lo = *(const v8h*)&stage[wv][nl][kt * 32 + koff];
      v8h hi = *(const v8h*)&stage[wv][nl][kt * 32 + koff + 16];
      a1[m][kt] = __builtin_shufflevector(lo, hi, 0,1,2,3,4,5,6,7,8,9,10,11,12,13,14,15);
    }
  }

  // ---- layer 1: 16x64 @ 64x128 (8 nt x 2 kt x MT WMMAs)
  #pragma unroll
  for (int nt = 0; nt < 8; ++nt) {
    const _Float16* fb = frags + F0_HALFS + (nt * 2) * 512 + lane * 16;
    #pragma unroll
    for (int kt = 0; kt < 2; ++kt) {
      v16h bf = *(const v16h*)(fb + kt * 512);
      #pragma unroll
      for (int m = 0; m < MT; ++m)
        acc[m][nt] = __builtin_amdgcn_wmma_f32_16x16x32_f16(
            false, a1[m][kt], false, bf, (short)0, kt ? acc[m][nt] : z, false, false);
    }
  }

  // stage h1 per M-tile, reload layer-2 A frags
  v16h a2[MT][4];
  #pragma unroll
  for (int m = 0; m < MT; ++m) {
    #pragma unroll
    for (int nt = 0; nt < 8; ++nt) {
      float bias = biases[64 + nt * 16 + nl];
      int nn = nt * 16 + nl;
      #pragma unroll
      for (int r = 0; r < 8; ++r) {
        float val = acc[m][nt][r] + bias;
        val = val > 0.f ? val : 0.f;
        stage[wv][mbase + r][nn] = (_Float16)val;
      }
    }
    asm volatile("s_wait_dscnt 0" ::: "memory");
    #pragma unroll
    for (int kt = 0; kt < 4; ++kt) {
      v8h lo = *(const v8h*)&stage[wv][nl][kt * 32 + koff];
      v8h hi = *(const v8h*)&stage[wv][nl][kt * 32 + koff + 16];
      a2[m][kt] = __builtin_shufflevector(lo, hi, 0,1,2,3,4,5,6,7,8,9,10,11,12,13,14,15);
    }
  }

  // ---- layer 2: 16x128 @ 128x128 (8 nt x 4 kt x MT WMMAs)
  #pragma unroll
  for (int nt = 0; nt < 8; ++nt) {
    const _Float16* fb = frags + F0_HALFS + F1_HALFS + (nt * 4) * 512 + lane * 16;
    #pragma unroll
    for (int kt = 0; kt < 4; ++kt) {
      v16h bf = *(const v16h*)(fb + kt * 512);
      #pragma unroll
      for (int m = 0; m < MT; ++m)
        acc[m][nt] = __builtin_amdgcn_wmma_f32_16x16x32_f16(
            false, a2[m][kt], false, bf, (short)0, kt ? acc[m][nt] : z, false, false);
    }
  }

  // ---- bias + ReLU + scatter-max (u32 bit-pattern max == float max for >=0)
  unsigned int* outu = (unsigned int*)out;
  #pragma unroll
  for (int m = 0; m < MT; ++m) {
    #pragma unroll
    for (int nt = 0; nt < 8; ++nt) {
      float bias = biases[192 + nt * 16 + nl];
      int nn = nt * 16 + nl;
      unsigned int choff = (unsigned int)nn * HW;
      #pragma unroll
      for (int r = 0; r < 8; ++r) {
        float val = acc[m][nt][r] + bias;
        val = val > 0.f ? val : 0.f;
        int enc = sidx[wv][m * 16 + mbase + r];
        if (enc >= 0) {
          atomicMax(outu + (unsigned int)enc + choff, __float_as_uint(val));
        }
      }
    }
  }
}

// ---------------------------------------------------------------------------
extern "C" void kernel_launch(void* const* d_in, const int* in_sizes, int n_in,
                              void* d_out, int out_size, void* d_ws, size_t ws_size,
                              hipStream_t stream) {
  const float* points = (const float*)d_in[0];
  const float* w0 = (const float*)d_in[1];  const float* b0 = (const float*)d_in[2];
  const float* g0 = (const float*)d_in[3];  const float* be0 = (const float*)d_in[4];
  const float* m0 = (const float*)d_in[5];  const float* v0 = (const float*)d_in[6];
  const float* w1 = (const float*)d_in[7];  const float* b1 = (const float*)d_in[8];
  const float* g1 = (const float*)d_in[9];  const float* be1 = (const float*)d_in[10];
  const float* m1 = (const float*)d_in[11]; const float* v1 = (const float*)d_in[12];
  const float* w2 = (const float*)d_in[13]; const float* b2 = (const float*)d_in[14];
  const float* g2 = (const float*)d_in[15]; const float* be2 = (const float*)d_in[16];
  const float* m2 = (const float*)d_in[17]; const float* v2 = (const float*)d_in[18];

  _Float16* frags  = (_Float16*)d_ws;                          // 53248 B, fragment-swizzled f16
  float*    biases = (float*)((char*)d_ws + FT_HALFS * 2);     // 320 f32 effective biases

  // 1) fold BN + swizzle weights
  int prep_threads = FT_HALFS + 320;
  prep_kernel<<<(prep_threads + 255) / 256, 256, 0, stream>>>(
      w0, b0, g0, be0, m0, v0,
      w1, b1, g1, be1, m1, v1,
      w2, b2, g2, be2, m2, v2,
      frags, biases);

  // 2) zero the BEV grid (4*128*128*128 floats = 2M float4s)
  zero_kernel<<<(4 * 128 * 128 * 128 / 4) / 256, 256, 0, stream>>>((float4*)d_out);

  // 3) MLP (WMMA) + scatter-max: 15000 macro-tiles of 32 points, 8 waves/block
  pillar_kernel<<<1875, 256, 0, stream>>>(points, frags, biases, (float*)d_out);
}
